// GNNDrugInteractionModel_7851200217342
// MI455X (gfx1250) — matrix operand order repaired and checked by reading, not compiled, not run on hardware
//
#include <hip/hip_runtime.h>
#include <hip/hip_bf16.h>

typedef __attribute__((ext_vector_type(16))) _Float16 v16h;
typedef __attribute__((ext_vector_type(8)))  float    v8f;

#define FDIM 128   // node feature dim == hidden dim
#define NODES_PER_SEG 32

// ---------------------------------------------------------------------------
// Elementwise conversion: f32 -> f16
__global__ void k_cvt_f16(const float* __restrict__ x, _Float16* __restrict__ xh, int n) {
    int i = blockIdx.x * blockDim.x + threadIdx.x;
    if (i < n) xh[i] = (_Float16)x[i];
}

// Transpose weight [K,N] f32 -> [N,K] f16 (B-fragment friendly: contiguous K per column)
__global__ void k_transpose_f16(const float* __restrict__ W, _Float16* __restrict__ Wt, int K, int N) {
    int i = blockIdx.x * blockDim.x + threadIdx.x;
    if (i >= K * N) return;
    int k = i / N, n = i % N;
    Wt[(size_t)n * K + k] = (_Float16)W[i];
}

// ---------------------------------------------------------------------------
// Per-node segment id: seg = 2*g + (local >= split[g]), g = searchsorted(ptr, i) - 1
__global__ void k_seg(const int* __restrict__ ptr, const int* __restrict__ split,
                      int* __restrict__ seg, int nNodes, int nB) {
    int i = blockIdx.x * blockDim.x + threadIdx.x;
    if (i >= nNodes) return;
    int lo = 0, hi = nB - 1;
    while (lo < hi) {
        int mid = (lo + hi + 1) >> 1;
        if (ptr[mid] <= i) lo = mid; else hi = mid - 1;
    }
    int local = i - ptr[lo];
    seg[i] = 2 * lo + (local >= split[lo] ? 1 : 0);
}

__global__ void k_deg_init(float* __restrict__ deg, int n) {
    int i = blockIdx.x * blockDim.x + threadIdx.x;
    if (i < n) deg[i] = 1.0f;   // self-loop
}

__global__ void k_deg_edges(const int* __restrict__ ei, const int* __restrict__ seg,
                            float* __restrict__ deg, int E) {
    int e = blockIdx.x * blockDim.x + threadIdx.x;
    if (e >= E) return;
    int s = ei[e], d = ei[E + e];
    if (seg[s] == seg[d]) atomicAdd(&deg[d], 1.0f);
}

__global__ void k_rsqrt(float* __restrict__ deg, int n) {
    int i = blockIdx.x * blockDim.x + threadIdx.x;
    if (i < n) deg[i] = rsqrtf(deg[i]);   // in-place deg -> dinv
}

// ---------------------------------------------------------------------------
// WMMA GEMM: C[M,N] (f32) = A[M,K] (f16 row-major) x Bt[N,K] (f16, pre-transposed)
// One wave per 16x16 output tile; K stepped by 32 via v_wmma_f32_16x16x32_f16.
__global__ void k_wmma_gemm(const _Float16* __restrict__ A, const _Float16* __restrict__ Bt,
                            float* __restrict__ C, int M, int N, int K) {
    int lane  = threadIdx.x & 31;
    int wave  = threadIdx.x >> 5;
    int wpb   = blockDim.x >> 5;
    int tile  = blockIdx.x * wpb + wave;
    int tilesN = N >> 4;
    int tm = (tile / tilesN) << 4;
    int tn = (tile % tilesN) << 4;
    if (tm >= M) return;

    int h16 = lane >> 4;     // 0: lanes 0-15, 1: lanes 16-31
    int l   = lane & 15;

    const _Float16* arow = A  + (size_t)(tm + l) * K;   // A row for M = tm+l
    const _Float16* bcol = Bt + (size_t)(tn + l) * K;   // B column for N = tn+l

    v8f acc = {};
    for (int k0 = 0; k0 < K; k0 += 32) {
        // A fragment (16x32 f16): lane half selects K group base (+0 / +8);
        // elements 0..7 <- K ka..ka+7, elements 8..15 <- K ka+16..ka+23
        v16h a, b;
        int ka = k0 + h16 * 8;
        float4* ap = reinterpret_cast<float4*>(&a);
        ap[0] = *reinterpret_cast<const float4*>(arow + ka);
        ap[1] = *reinterpret_cast<const float4*>(arow + ka + 16);
        // B fragment (32x16 f16): lane half selects K group base (+0 / +8);
        // elements 0..7 <- K kb..kb+7, elements 8..15 <- K kb+16..kb+23 of column tn+l
        int kb = k0 + h16 * 8;
        float4* bp = reinterpret_cast<float4*>(&b);
        bp[0] = *reinterpret_cast<const float4*>(bcol + kb);
        bp[1] = *reinterpret_cast<const float4*>(bcol + kb + 16);

        acc = __builtin_amdgcn_wmma_f32_16x16x32_f16(
            /*neg_a=*/false, a, /*neg_b=*/false, b,
            /*c_mod=*/(short)0, acc, /*reuse_a=*/false, /*reuse_b=*/false);
    }

    // C/D layout: VGPR r -> row r (lanes 0-15) / row r+8 (lanes 16-31), col = lane&15
    float* crow = C + (size_t)(tm + h16 * 8) * N + tn + l;
    #pragma unroll
    for (int r = 0; r < 8; ++r)
        crow[(size_t)r * N] = acc[r];
}

// ---------------------------------------------------------------------------
// agg[i,f] = dinv[i]^2 * hw[i,f] + b[f]   (self-loop term + bias, float4 per thread)
__global__ void k_agg_init(const float* __restrict__ hw, const float* __restrict__ dinv,
                           const float* __restrict__ bias, float* __restrict__ agg, int nNodes) {
    int t = blockIdx.x * blockDim.x + threadIdx.x;     // nNodes * 32 threads
    if (t >= nNodes * (FDIM / 4)) return;
    int node = t >> 5, q = t & 31;
    float sn = dinv[node] * dinv[node];
    float4 h = *reinterpret_cast<const float4*>(hw + (size_t)node * FDIM + q * 4);
    float4 b = *reinterpret_cast<const float4*>(bias + q * 4);
    float4 o = make_float4(sn * h.x + b.x, sn * h.y + b.y, sn * h.z + b.z, sn * h.w + b.w);
    *reinterpret_cast<float4*>(agg + (size_t)node * FDIM + q * 4) = o;
}

// Edge aggregation: one wave per edge, 4 features per lane.
// agg[dst] += dinv[src]*dinv[dst] * hw[src]   (only if same segment)
__global__ void k_edge_agg(const int* __restrict__ ei, const int* __restrict__ seg,
                           const float* __restrict__ dinv, const float* __restrict__ hw,
                           float* __restrict__ agg, int E) {
    long long gid = (long long)blockIdx.x * blockDim.x + threadIdx.x;
    int e = (int)(gid >> 5), lane = (int)(gid & 31);
    if (e >= E) return;
    int s = ei[e], d = ei[E + e];
    if (seg[s] != seg[d]) return;
    float nm = dinv[s] * dinv[d];
    float4 h = *reinterpret_cast<const float4*>(hw + (size_t)s * FDIM + lane * 4);
    float* dp = agg + (size_t)d * FDIM + lane * 4;
    atomicAdd(dp + 0, nm * h.x);
    atomicAdd(dp + 1, nm * h.y);
    atomicAdd(dp + 2, nm * h.z);
    atomicAdd(dp + 3, nm * h.w);
}

// ReLU + convert to f16 (feeds next WMMA layer as A-matrix)
__global__ void k_relu_f16(const float* __restrict__ agg, _Float16* __restrict__ out, int n) {
    int i = blockIdx.x * blockDim.x + threadIdx.x;
    if (i < n) out[i] = (_Float16)fmaxf(agg[i], 0.0f);
}

// ---------------------------------------------------------------------------
// Mean pool (with fused ReLU) over 32 contiguous nodes per segment; write
// directly into the interleaved pair layout [B, 2H] as f16 for the MLP GEMM.
__global__ void k_pool(const float* __restrict__ agg2, _Float16* __restrict__ pairh, int nSeg) {
    int s = blockIdx.x;            // one block per segment
    int f = threadIdx.x;           // 128 features
    if (s >= nSeg) return;
    const float* base = agg2 + (size_t)s * NODES_PER_SEG * FDIM + f;
    float sum = 0.0f;
    #pragma unroll
    for (int i = 0; i < NODES_PER_SEG; ++i)
        sum += fmaxf(base[(size_t)i * FDIM], 0.0f);
    sum *= (1.0f / NODES_PER_SEG);
    int b = s >> 1, hsel = s & 1;
    pairh[(size_t)b * (2 * FDIM) + hsel * FDIM + f] = (_Float16)sum;
}

// Final: relu(z1 + bm1) . Wm2 + bm2 -> sigmoid. One wave per row, shfl reduce.
__global__ void k_mlp_out(const float* __restrict__ z1, const float* __restrict__ bm1,
                          const float* __restrict__ Wm2, const float* __restrict__ bm2,
                          float* __restrict__ out, int rows) {
    int gid = blockIdx.x * blockDim.x + threadIdx.x;
    int row = gid >> 5, lane = gid & 31;
    if (row >= rows) return;
    float acc = 0.0f;
    #pragma unroll
    for (int j = lane; j < FDIM; j += 32) {
        float z = fmaxf(z1[(size_t)row * FDIM + j] + bm1[j], 0.0f);
        acc += z * Wm2[j];
    }
    for (int off = 16; off >= 1; off >>= 1)
        acc += __shfl_down(acc, off, 32);
    if (lane == 0) {
        float v = acc + bm2[0];
        out[row] = 1.0f / (1.0f + __expf(-v));
    }
}

// ---------------------------------------------------------------------------
static inline size_t alignup(size_t x) { return (x + 255) & ~(size_t)255; }

extern "C" void kernel_launch(void* const* d_in, const int* in_sizes, int n_in,
                              void* d_out, int out_size, void* d_ws, size_t ws_size,
                              hipStream_t stream) {
    const float* x    = (const float*)d_in[0];
    const int*   ei   = (const int*)d_in[1];
    const int*   ptr  = (const int*)d_in[2];
    const int*   split= (const int*)d_in[3];
    const float* W1   = (const float*)d_in[4];
    const float* b1   = (const float*)d_in[5];
    const float* W2   = (const float*)d_in[6];
    const float* b2   = (const float*)d_in[7];
    const float* Wm1  = (const float*)d_in[8];
    const float* bm1  = (const float*)d_in[9];
    const float* Wm2  = (const float*)d_in[10];
    const float* bm2  = (const float*)d_in[11];
    float* out = (float*)d_out;

    const int nNodes = in_sizes[0] / FDIM;        // 131072
    const int E      = in_sizes[1] / 2;           // 1048576
    const int nB     = in_sizes[2] - 1;           // 2048
    const int nSeg   = 2 * nB;                    // 4096
    const int K2H    = 2 * FDIM;                  // 256
    const int NF     = nNodes * FDIM;

    // Workspace carve-up
    char* ws = (char*)d_ws;
    size_t o = 0;
    _Float16* xh   = (_Float16*)(ws + o); o += alignup((size_t)NF * 2);        // also reused as h1h
    float*    hw   = (float*)   (ws + o); o += alignup((size_t)NF * 4);
    float*    agg  = (float*)   (ws + o); o += alignup((size_t)NF * 4);
    int*      seg  = (int*)     (ws + o); o += alignup((size_t)nNodes * 4);
    float*    dinv = (float*)   (ws + o); o += alignup((size_t)nNodes * 4);
    _Float16* W1t  = (_Float16*)(ws + o); o += alignup((size_t)FDIM * FDIM * 2);
    _Float16* W2t  = (_Float16*)(ws + o); o += alignup((size_t)FDIM * FDIM * 2);
    _Float16* Wm1t = (_Float16*)(ws + o); o += alignup((size_t)K2H * FDIM * 2);
    _Float16* pairh= (_Float16*)(ws + o); o += alignup((size_t)nB * K2H * 2);
    float*    z1   = (float*)   (ws + o); o += alignup((size_t)nB * FDIM * 4);
    (void)ws_size;

    const int T = 256;
    #define BLK(n) (((n) + T - 1) / T)

    // --- preprocessing ---
    k_cvt_f16      <<<BLK(NF), T, 0, stream>>>(x, xh, NF);
    k_transpose_f16<<<BLK(FDIM*FDIM), T, 0, stream>>>(W1, W1t, FDIM, FDIM);
    k_transpose_f16<<<BLK(FDIM*FDIM), T, 0, stream>>>(W2, W2t, FDIM, FDIM);
    k_transpose_f16<<<BLK(K2H*FDIM),  T, 0, stream>>>(Wm1, Wm1t, K2H, FDIM);
    k_seg          <<<BLK(nNodes), T, 0, stream>>>(ptr, split, seg, nNodes, nB);
    k_deg_init     <<<BLK(nNodes), T, 0, stream>>>(dinv, nNodes);
    k_deg_edges    <<<BLK(E), T, 0, stream>>>(ei, seg, dinv, E);
    k_rsqrt        <<<BLK(nNodes), T, 0, stream>>>(dinv, nNodes);

    const int gemmTiles1 = (nNodes / 16) * (FDIM / 16);
    const long long edgeThreads = (long long)E * 32;

    // --- GCN layer 1: hw = xh @ W1 ; agg = selfnorm*hw + b1 ; agg += edges ; h1 = relu ---
    k_wmma_gemm <<<gemmTiles1 / 8, 256, 0, stream>>>(xh, W1t, hw, nNodes, FDIM, FDIM);
    k_agg_init  <<<BLK(nNodes * 32), T, 0, stream>>>(hw, dinv, b1, agg, nNodes);
    k_edge_agg  <<<(int)(edgeThreads / T), T, 0, stream>>>(ei, seg, dinv, hw, agg, E);
    k_relu_f16  <<<BLK(NF), T, 0, stream>>>(agg, xh, NF);          // reuse xh as h1 (f16)

    // --- GCN layer 2 ---
    k_wmma_gemm <<<gemmTiles1 / 8, 256, 0, stream>>>(xh, W2t, hw, nNodes, FDIM, FDIM);
    k_agg_init  <<<BLK(nNodes * 32), T, 0, stream>>>(hw, dinv, b2, agg, nNodes);
    k_edge_agg  <<<(int)(edgeThreads / T), T, 0, stream>>>(ei, seg, dinv, hw, agg, E);

    // --- pooling (fused ReLU) -> pair layout f16 ---
    k_pool <<<nSeg, FDIM, 0, stream>>>(agg, pairh, nSeg);

    // --- MLP: z1 = pair @ Wm1 (WMMA, K=256) ; out = sigmoid(relu(z1+bm1)@Wm2+bm2) ---
    const int gemmTiles3 = (nB / 16) * (FDIM / 16);
    k_wmma_gemm <<<gemmTiles3 / 8, 256, 0, stream>>>(pairh, Wm1t, z1, nB, FDIM, K2H);
    k_mlp_out   <<<BLK(nB * 32), T, 0, stream>>>(z1, bm1, Wm2, bm2, out, nB);

    #undef BLK
}
